// sLSTMLayer_69698729279879
// MI455X (gfx1250) — matrix operand order repaired and checked by reading, not compiled, not run on hardware
//
#include <hip/hip_runtime.h>
#include <cstdint>
#include <cstddef>

// ---------------- problem constants ----------------
#define BB     4096
#define HH     2048
#define NHEAD  8
#define DHEAD  256
#define KCONV  4

typedef __bf16 bf16_t;
typedef __attribute__((ext_vector_type(16))) __bf16 v16bf;
typedef __attribute__((ext_vector_type(8)))  float  v8f;

union FragBF { v16bf v; uint4 u[2]; };

// =====================================================================
// Kernel 1: repack weights (Wi,Wf,Wz,Wo,Ri,Rf,Rz,Ro) into WMMA-B-frag
// native bf16 hi/lo arrays.  Packed element index bit layout:
//   g(3) | head(3) | kstep(3) | ntile(4) | lane(5) | j(4)   (22 bits = 4M)
// B fragment for (kstep,ntile): lane reads 16 contiguous bf16
// (= two coalesced uint4 loads).  Element = W[head][k][n] with
//   k = kstep*32 + (lane>>4)*16 + j ,  n = ntile*16 + (lane&15)
// a contiguous k order, the same one the A fragment loads use.
// =====================================================================
__global__ __launch_bounds__(256, 1) void slstm_pack(
    const float* __restrict__ w0, const float* __restrict__ w1,
    const float* __restrict__ w2, const float* __restrict__ w3,
    const float* __restrict__ w4, const float* __restrict__ w5,
    const float* __restrict__ w6, const float* __restrict__ w7,
    bf16_t* __restrict__ hi, bf16_t* __restrict__ lo)
{
    int t = blockIdx.x * 256 + threadIdx.x;     // < 4,194,304
    int j     =  t        & 15;
    int lane  = (t >> 4)  & 31;
    int nt    = (t >> 9)  & 15;
    int kstep = (t >> 13) & 7;
    int head  = (t >> 16) & 7;
    int g     = (t >> 19) & 7;
    const float* W;
    switch (g) {
        case 0: W = w0; break; case 1: W = w1; break;
        case 2: W = w2; break; case 3: W = w3; break;
        case 4: W = w4; break; case 5: W = w5; break;
        case 6: W = w6; break; default: W = w7; break;
    }
    int k = kstep * 32 + (lane >> 4) * 16 + j;
    int n = nt * 16 + (lane & 15);
    float v = W[((size_t)head * 256 + k) * 256 + n];
    bf16_t hv = (bf16_t)v;
    hi[t] = hv;
    lo[t] = (bf16_t)(v - (float)hv);
}

// =====================================================================
// Kernel 2: causal depthwise conv (single step) + SiLU + state shift.
// =====================================================================
__global__ __launch_bounds__(256, 1) void slstm_conv(
    const float* __restrict__ xin,        // (B,H)
    const float* __restrict__ cstate,     // (B,K,H)
    const float* __restrict__ ck,         // (K,H)
    const float* __restrict__ cb,         // (H)
    float* __restrict__ conv_act,         // (B,H) ws
    float* __restrict__ out_cs)           // (B,K,H) -> d_out region 5
{
    size_t i = (size_t)blockIdx.x * 256 + threadIdx.x;   // < B*H
    int b  = (int)(i >> 11);          // /H  (H=2048)
    int hc = (int)(i & (HH - 1));
    size_t sb = (size_t)b * KCONV * HH + hc;
    float w0 = cstate[sb + 1 * HH];
    float w1 = cstate[sb + 2 * HH];
    float w2 = cstate[sb + 3 * HH];
    float w3 = xin[i];
    float x = w0 * ck[0 * HH + hc] + w1 * ck[1 * HH + hc]
            + w2 * ck[2 * HH + hc] + w3 * ck[3 * HH + hc] + cb[hc];
    conv_act[i] = x / (1.0f + expf(-x));                  // SiLU
    out_cs[sb + 0 * HH] = w0;
    out_cs[sb + 1 * HH] = w1;
    out_cs[sb + 2 * HH] = w2;
    out_cs[sb + 3 * HH] = w3;
}

// =====================================================================
// Kernel 3: flag = (any n != 0)   (for the jnp.all(n==0) broadcast)
// =====================================================================
__global__ __launch_bounds__(256, 1) void slstm_nzflag(
    const float* __restrict__ n, int* __restrict__ flag)
{
    __shared__ int found;
    if (threadIdx.x == 0) found = 0;
    __syncthreads();
    size_t base = (size_t)blockIdx.x * 256 * 8 + threadIdx.x;
    bool nz = false;
    #pragma unroll
    for (int q = 0; q < 8; ++q)
        if (n[base + (size_t)q * 256] != 0.0f) nz = true;
    if (nz) found = 1;
    __syncthreads();
    if (threadIdx.x == 0 && found) atomicOr(flag, 1);
}

// =====================================================================
// Kernel 4: fused block-diag GEMMs (bf16x3 split WMMA, fp32 accum)
//           + exp-gating + state update + per-head LayerNorm.
//
// Spill-free: n-tile is the OUTER loop; all K=512 of activations for
// the block's 16 rows stay resident in LDS, so only two 8-VGPR
// accumulator tiles are live.  Two accumulators (hi*hi vs cross terms)
// break the WMMA->WMMA RAW chain (distance 1 -> 2..3).
//
// Grid: (B/16, NH); block: 256 threads = 8 waves.
// Wave w: gate = w&3 (i,f,z,o), ch = w>>2 (column half, 8 n-tiles).
// =====================================================================
__global__ __launch_bounds__(256, 1) void slstm_fused_gemm(
    const float* __restrict__ conv_act, const float* __restrict__ xin,
    const float* __restrict__ hprev,
    const bf16_t* __restrict__ pHi, const bf16_t* __restrict__ pLo,
    const float* __restrict__ ibias, const float* __restrict__ fbias,
    const float* __restrict__ zbias, const float* __restrict__ obias,
    const float* __restrict__ cprev, const float* __restrict__ nprev,
    const float* __restrict__ mprev, const float* __restrict__ lnscale,
    const int* __restrict__ nz_flag,
    float* __restrict__ y_out, float* __restrict__ c_out,
    float* __restrict__ n_out, float* __restrict__ m_out,
    float* __restrict__ h_out)
{
    // LDS: [0,24K)      sHi[3][16][256] bf16   (conv_act, inputs, h)
    //      [24K,48K)    sLo[3][16][256] bf16
    //      [48K,56K)    xchg[4][16][32] f32    gate slab exchange
    //      [56K,+128)   mus[16][2] f32
    __shared__ __align__(16) unsigned char smem[57472];
    bf16_t* sHi  = (bf16_t*)smem;
    bf16_t* sLo  = (bf16_t*)(smem + 24576);
    float*  xchg = (float*)(smem + 49152);
    float*  mus  = (float*)(smem + 57344);

    const int tid     = threadIdx.x;
    const int lane    = tid & 31;
    const int wid     = tid >> 5;
    const int gate    = wid & 3;          // 0:i 1:f 2:z 3:o
    const int ch      = wid >> 2;         // column half (0..1)
    const int head    = blockIdx.y;
    const int rowbase = blockIdx.x * 16;
    const int m16     = lane & 15;
    const int kh      = lane >> 4;

    // ---- stage all activations (bf16 hi/lo split) once: 48KB ----
    for (int e = tid; e < 3 * 16 * 256; e += 256) {
        int srcI = e >> 12;               // /(16*256)
        int rem  = e & 4095;
        int row  = rem >> 8, col = rem & 255;
        const float* src = (srcI == 0) ? conv_act : (srcI == 1) ? xin : hprev;
        float v = src[(size_t)(rowbase + row) * HH + head * DHEAD + col];
        bf16_t hv = (bf16_t)v;
        sHi[(srcI * 16 + row) * 256 + col] = hv;
        sLo[(srcI * 16 + row) * 256 + col] = (bf16_t)(v - (float)hv);
    }
    __syncthreads();

    const bool allzero = (*nz_flag == 0);
    const int  srcX    = (gate >= 2) ? 1 : 0;   // i,f: conv_act  z,o: inputs
    // LN partials: this wave owns rows wid and wid+8 in the elementwise map
    float sum0 = 0.0f, sq0 = 0.0f, sum1 = 0.0f, sq1 = 0.0f;

    for (int ntl = 0; ntl < 8; ++ntl) {
        const int ntg = ch * 8 + ntl;           // global n-tile 0..15
        v8f accA = {};                          // hi*hi term
        v8f accB = {};                          // cross terms

        #pragma unroll
        for (int phase = 0; phase < 2; ++phase) {
            const int pg     = phase ? 4 + gate : gate;       // W vs R
            const int srcIdx = phase ? 2 : srcX;
            const bf16_t* WH = pHi + (size_t)(pg * NHEAD + head) * 65536;
            const bf16_t* WL = pLo + (size_t)(pg * NHEAD + head) * 65536;
            const int abase  = (srcIdx * 16 + m16) * 256 + kh * 16;

            #pragma unroll
            for (int kstep = 0; kstep < 8; ++kstep) {
                // A fragment (16x32 bf16): two b128 LDS loads per hi/lo
                FragBF AH, AL;
                const uint4* pa = (const uint4*)&sHi[abase + kstep * 32];
                AH.u[0] = pa[0]; AH.u[1] = pa[1];
                const uint4* pl = (const uint4*)&sLo[abase + kstep * 32];
                AL.u[0] = pl[0]; AL.u[1] = pl[1];
                // B fragment: 32B contiguous per lane, coalesced
                const size_t boff = (size_t)(kstep * 16 + ntg) * 512 + (size_t)lane * 16;
                FragBF BH, BL;
                const uint4* pb  = (const uint4*)(WH + boff);
                BH.u[0] = pb[0];  BH.u[1] = pb[1];
                const uint4* pbl = (const uint4*)(WL + boff);
                BL.u[0] = pbl[0]; BL.u[1] = pbl[1];
                // bf16x3 split product, fp32 accumulate, 2 dep chains
                accA = __builtin_amdgcn_wmma_f32_16x16x32_bf16(
                    false, AH.v, false, BH.v, (short)0, accA, false, false);
                accB = __builtin_amdgcn_wmma_f32_16x16x32_bf16(
                    false, AH.v, false, BL.v, (short)0, accB, false, false);
                accB = __builtin_amdgcn_wmma_f32_16x16x32_bf16(
                    false, AL.v, false, BH.v, (short)0, accB, false, false);
            }
        }
        v8f acc = accA + accB;

        // ---- exchange the 4 gates for this slab pair through LDS ----
        __syncthreads();        // previous iteration's elementwise done
        #pragma unroll
        for (int vr = 0; vr < 8; ++vr) {            // C layout: row=kh*8+vr, col=m16
            xchg[(gate * 16 + kh * 8 + vr) * 32 + ch * 16 + m16] = acc[vr];
        }
        __syncthreads();

        // ---- fused gating for 16 rows x 32 cols (both column halves) ----
        #pragma unroll
        for (int q = 0; q < 2; ++q) {
            int e     = tid + q * 256;
            int row   = e >> 5;                     // 0..15
            int cc    = e & 31;
            int chx   = cc >> 4;
            int col16 = cc & 15;
            int colg  = chx * 128 + ntl * 16 + col16;
            int hcol  = head * DHEAD + colg;
            size_t gi = (size_t)(rowbase + row) * HH + hcol;

            float xi = xchg[(0 * 16 + row) * 32 + cc] + ibias[hcol];
            float xf = xchg[(1 * 16 + row) * 32 + cc] + fbias[hcol];
            float xz = xchg[(2 * 16 + row) * 32 + cc] + zbias[hcol];
            float xo = xchg[(3 * 16 + row) * 32 + cc] + obias[hcol];

            float osig = 1.0f / (1.0f + expf(-xo));
            // log_f = -softplus(-xf)  (stable)
            float lgf  = -(fmaxf(-xf, 0.0f) + log1pf(expf(-fabsf(xf))));
            float mold = mprev[gi];
            float mnew = allzero ? xi : fmaxf(lgf + mold, xi);
            float ip   = fminf(expf(xi - mnew), 1.0f);
            float fp   = fminf(expf(lgf + mold - mnew), 1.0f);
            float cn   = fp * cprev[gi] + ip * tanhf(xz);
            float nn   = fp * nprev[gi] + ip;
            float hn   = osig * (cn / fmaxf(nn, 1e-6f));

            c_out[gi] = cn;  n_out[gi] = nn;  m_out[gi] = mnew;  h_out[gi] = hn;
            if (q == 0) { sum0 += hn; sq0 += hn * hn; }
            else        { sum1 += hn; sq1 += hn * hn; }
        }
    }

    // ---- per-head LayerNorm stats: full-wave reduction ----
    // wave w holds partials for rows w (sum0) and w+8 (sum1)
    #pragma unroll
    for (int off = 1; off < 32; off <<= 1) {
        sum0 += __shfl_xor(sum0, off, 32);
        sq0  += __shfl_xor(sq0,  off, 32);
        sum1 += __shfl_xor(sum1, off, 32);
        sq1  += __shfl_xor(sq1,  off, 32);
    }
    if (lane == 0) {
        float mu0 = sum0 * (1.0f / 256.0f);
        float mu1 = sum1 * (1.0f / 256.0f);
        mus[wid * 2]            = mu0;
        mus[wid * 2 + 1]        = rsqrtf(sq0 * (1.0f / 256.0f) - mu0 * mu0 + 1e-6f);
        mus[(wid + 8) * 2]      = mu1;
        mus[(wid + 8) * 2 + 1]  = rsqrtf(sq1 * (1.0f / 256.0f) - mu1 * mu1 + 1e-6f);
    }
    __threadfence();     // make h_out writes visible for the re-read below
    __syncthreads();

    // ---- LN apply: re-read h (L2-resident), write normalized output ----
    float gsc = lnscale[head * DHEAD + tid];     // tid == col (DH=256)
    for (int r = 0; r < 16; ++r) {
        size_t gi = (size_t)(rowbase + r) * HH + head * DHEAD + tid;
        float hv = h_out[gi];
        y_out[gi] = (hv - mus[r * 2]) * mus[r * 2 + 1] * gsc;
    }
}

// =====================================================================
// Host-side launcher
// =====================================================================
extern "C" void kernel_launch(void* const* d_in, const int* in_sizes, int n_in,
                              void* d_out, int out_size, void* d_ws, size_t ws_size,
                              hipStream_t stream)
{
    (void)in_sizes; (void)n_in; (void)out_size;

    const float* xin   = (const float*)d_in[0];
    const float* c_in  = (const float*)d_in[1];
    const float* n_in_ = (const float*)d_in[2];
    const float* m_in  = (const float*)d_in[3];
    const float* h_in  = (const float*)d_in[4];
    const float* cs_in = (const float*)d_in[5];
    const float* ck    = (const float*)d_in[6];
    const float* cb    = (const float*)d_in[7];
    const float* Wi = (const float*)d_in[8],  *Wf = (const float*)d_in[9];
    const float* Wz = (const float*)d_in[10], *Wo = (const float*)d_in[11];
    const float* Ri = (const float*)d_in[12], *Rf = (const float*)d_in[13];
    const float* Rz = (const float*)d_in[14], *Ro = (const float*)d_in[15];
    const float* ib = (const float*)d_in[16], *fb = (const float*)d_in[17];
    const float* zb = (const float*)d_in[18], *ob = (const float*)d_in[19];
    const float* ls = (const float*)d_in[20];

    const size_t BH = (size_t)BB * HH;            // 8,388,608
    float* out_y  = (float*)d_out;
    float* out_c  = out_y + BH;
    float* out_n  = out_y + 2 * BH;
    float* out_m  = out_y + 3 * BH;
    float* out_h  = out_y + 4 * BH;
    float* out_cs = out_y + 5 * BH;               // (B,K,H)

    // workspace: packedHi 8MB | packedLo 8MB | conv_act 32MB | flag
    unsigned char* ws = (unsigned char*)d_ws;
    bf16_t* pHi      = (bf16_t*)ws;
    bf16_t* pLo      = (bf16_t*)(ws + 8388608);
    float*  conv_act = (float*)(ws + 16777216);
    int*    flag     = (int*)(ws + 50331648);
    (void)ws_size;

    hipMemsetAsync(flag, 0, sizeof(int), stream);

    slstm_pack<<<16384, 256, 0, stream>>>(Wi, Wf, Wz, Wo, Ri, Rf, Rz, Ro, pHi, pLo);
    slstm_conv<<<(unsigned)(BH / 256), 256, 0, stream>>>(xin, cs_in, ck, cb, conv_act, out_cs);
    slstm_nzflag<<<(unsigned)(BH / (256 * 8)), 256, 0, stream>>>(n_in_, flag);

    dim3 grid(BB / 16, NHEAD);
    slstm_fused_gemm<<<grid, 256, 0, stream>>>(
        conv_act, xin, h_in, pHi, pLo,
        ib, fb, zb, ob, c_in, n_in_, m_in, ls, flag,
        out_y, out_c, out_n, out_m, out_h);
}
